// MultiHeadAttention_69329362092168
// MI455X (gfx1250) — matrix operand order repaired and checked by reading, not compiled, not run on hardware
//
#include <hip/hip_runtime.h>
#include <hip/hip_bf16.h>

typedef __attribute__((ext_vector_type(16))) __bf16 v16bf;
typedef __attribute__((ext_vector_type(8)))  float  v8f;

#define AS1 __attribute__((address_space(1)))
#define AS3 __attribute__((address_space(3)))

#if defined(__has_builtin)
#if __has_builtin(__builtin_amdgcn_global_load_async_to_lds_b128)
#define HAVE_ASYNC_LDS 1
#endif
#endif

#if HAVE_ASYNC_LDS
typedef int v4i_vs __attribute__((vector_size(16)));   // matches builtin param
__device__ __forceinline__ void async_copy16(const void* gsrc, void* ldst) {
    __builtin_amdgcn_global_load_async_to_lds_b128(
        (AS1 v4i_vs*)gsrc, (AS3 v4i_vs*)ldst, 0, 0);
}
#endif

union ABFrag {
    v16bf v;
    unsigned short h[16];
    unsigned int   u[8];
};

__device__ __forceinline__ unsigned short f2bf(float f) {
    union { float f; unsigned u; } x; x.f = f;
    unsigned r = x.u + 0x7FFFu + ((x.u >> 16) & 1u);   // round-to-nearest-even
    return (unsigned short)(r >> 16);
}

// A-fragment (16x32 bf16) pair base K index for vgpr v, lane-half h
__device__ __forceinline__ int kA(int v, int half) {
    return (v < 4 ? 2 * v : 16 + 2 * (v - 4)) + 8 * half;
}
// B-fragment (32x16 bf16) pair base K index for vgpr r, lane-half h
__device__ __forceinline__ int kB(int r, int half) {
    return (r < 4 ? 2 * r : 16 + 2 * (r - 4)) + 8 * half;
}

// ---------------------------------------------------------------------------
// Tiled GEMM with bias:  C[M,N] = A[M,K] * W[K,N] + bias
// AMODE: 0 = A is f32, 1 = A is bf16 (ushort)
// SMODE: 0 = store f32 row-major [M,N]
//        1 = store bf16 head-split [B,H,T,DH]  (N == H*DH, M == B*T)
// Block: 128 threads (4 waves). Block tile 64x128, wave tile 32x64 (2x4 WMMA).
// ---------------------------------------------------------------------------
template <int AMODE, int SMODE>
__global__ __launch_bounds__(128) void gemm_bias_kernel(
    const void* __restrict__ Aptr, const float* __restrict__ W,
    const float* __restrict__ bias, void* __restrict__ Cptr,
    int M, int N, int K, int T_, int H_, int DH_)
{
    __shared__ unsigned short lA[64 * 32];    // [m][k] bf16
    __shared__ unsigned short lWT[128 * 32];  // [n][k] bf16 (W tile transposed)

    const int tid  = threadIdx.x;
    const int lane = tid & 31;
    const int wave = tid >> 5;
    const int half = lane >> 4;
    const int l16  = lane & 15;

    const int m0 = blockIdx.y * 64;
    const int n0 = blockIdx.x * 128;
    const int wm = (wave >> 1) * 32;
    const int wn = (wave & 1) * 64;

    v8f acc[2][4] = {};

    for (int kt = 0; kt < K; kt += 32) {
        // ---- stage A tile (64 x 32) as bf16 ----
        if (AMODE == 0) {
            const float* A = (const float*)Aptr;
            for (int i = tid; i < 64 * 8; i += 128) {
                int r = i >> 3, c4 = (i & 7) * 4;
                const float* src = A + (size_t)(m0 + r) * K + kt + c4;
                float4 f = *(const float4*)src;
                if (kt + 32 < K) __builtin_prefetch(src + 32, 0, 1);
                unsigned short* d = &lA[r * 32 + c4];
                d[0] = f2bf(f.x); d[1] = f2bf(f.y);
                d[2] = f2bf(f.z); d[3] = f2bf(f.w);
            }
        } else {
            const unsigned short* A = (const unsigned short*)Aptr;
            for (int i = tid; i < 64 * 4; i += 128) {
                int r = i >> 2, c8 = (i & 3) * 8;
                const ushort4* src =
                    (const ushort4*)(A + (size_t)(m0 + r) * K + kt + c8);
                if (kt + 32 < K) __builtin_prefetch(src + 4, 0, 1);
                ushort4 a = src[0], b = src[1];
                *(ushort4*)&lA[r * 32 + c8]     = a;
                *(ushort4*)&lA[r * 32 + c8 + 4] = b;
            }
        }
        // ---- stage W tile (32 x 128) transposed -> lWT[n][k] ----
        for (int i = tid; i < 32 * 32; i += 128) {
            int k = i >> 5, c4 = (i & 31) * 4;
            const float* src = W + (size_t)(kt + k) * N + n0 + c4;
            float4 f = *(const float4*)src;
            if (kt + 32 < K) __builtin_prefetch(src + (size_t)32 * N, 0, 1);
            lWT[(c4 + 0) * 32 + k] = f2bf(f.x);
            lWT[(c4 + 1) * 32 + k] = f2bf(f.y);
            lWT[(c4 + 2) * 32 + k] = f2bf(f.z);
            lWT[(c4 + 3) * 32 + k] = f2bf(f.w);
        }
        __syncthreads();

        // ---- build fragments and multiply ----
        ABFrag a[2], b[4];
#pragma unroll
        for (int t = 0; t < 2; ++t) {
            int row = wm + t * 16 + l16;
#pragma unroll
            for (int v = 0; v < 8; ++v)
                a[t].u[v] = *(const unsigned int*)&lA[row * 32 + kA(v, half)];
        }
#pragma unroll
        for (int t = 0; t < 4; ++t) {
            int col = wn + t * 16 + l16;
#pragma unroll
            for (int r = 0; r < 8; ++r)
                b[t].u[r] = *(const unsigned int*)&lWT[col * 32 + kB(r, half)];
        }
#pragma unroll
        for (int i = 0; i < 2; ++i)
#pragma unroll
            for (int j = 0; j < 4; ++j)
                acc[i][j] = __builtin_amdgcn_wmma_f32_16x16x32_bf16(
                    false, a[i].v, false, b[j].v, (short)0, acc[i][j],
                    false, false);
        __syncthreads();
    }

    // ---- epilogue: bias + store ----
#pragma unroll
    for (int i = 0; i < 2; ++i)
#pragma unroll
        for (int j = 0; j < 4; ++j) {
            int nCol = n0 + wn + j * 16 + l16;
            float bv = bias[nCol];
#pragma unroll
            for (int r = 0; r < 8; ++r) {
                int mRow = m0 + wm + i * 16 + r + 8 * half;
                float val = acc[i][j][r] + bv;
                if (SMODE == 0) {
                    ((float*)Cptr)[(size_t)mRow * N + nCol] = val;
                } else {
                    int bb = mRow / T_, t = mRow % T_;
                    int hh = nCol / DH_, d = nCol % DH_;
                    ((unsigned short*)Cptr)
                        [(((size_t)(bb * H_ + hh)) * T_ + t) * DH_ + d] =
                            f2bf(val);
                }
            }
        }
}

// ---------------------------------------------------------------------------
// Flash-attention (causal): Qh/Kh/Vh bf16 [B,H,T,DH] -> ctx bf16 [B,T,H*DH]
// Block: 128 threads (4 waves); each wave owns 16 query rows.
// Key loop is block-uniform so K/V tiles are staged cooperatively in LDS
// (K via async-to-LDS when available; V transposed through registers).
// ---------------------------------------------------------------------------
__global__ __launch_bounds__(128) void attn_kernel(
    const unsigned short* __restrict__ Qh, const unsigned short* __restrict__ Kh,
    const unsigned short* __restrict__ Vh, unsigned short* __restrict__ ctx,
    int T_, int Hc, int DHc)
{
    __shared__ unsigned short lK[32 * 64];    // [key][dh]  (async staged)
    __shared__ unsigned short lVT[64 * 32];   // [dh][key]  (register-transposed)
    __shared__ unsigned short lP[4][16 * 32]; // per-wave P re-layout staging

    const int tid  = threadIdx.x;
    const int lane = tid & 31;
    const int wave = tid >> 5;
    const int half = lane >> 4;
    const int l16  = lane & 15;

    const int bh  = blockIdx.y;                 // b*H + h
    const int b   = bh / Hc, h = bh % Hc;
    const int q0b = blockIdx.x * 64;            // block's first query row
    const int q0  = q0b + wave * 16;            // wave's first query row
    const size_t base = (size_t)bh * T_ * DHc;

    // ---- Q A-fragments (DH=64 -> two 16x32 fragments) ----
    ABFrag qa[2];
#pragma unroll
    for (int f = 0; f < 2; ++f)
#pragma unroll
        for (int v = 0; v < 8; ++v) {
            int dh = kA(v, half) + 32 * f;
            qa[f].u[v] = *(const unsigned int*)
                &Qh[base + (size_t)(q0 + l16) * DHc + dh];
        }

    v8f o[4] = {};              // 16x64 f32 output accumulator (4 n-chunks)
    float rowMax[8], rowSum[8];
#pragma unroll
    for (int r = 0; r < 8; ++r) { rowMax[r] = -1e30f; rowSum[r] = 0.f; }

    const float scale = 0.125f;    // 1/sqrt(DH=64)
    const int kEndBlk = q0b + 64;  // block-uniform causal bound

    for (int kt = 0; kt < kEndBlk; kt += 32) {
        // ---- stage K tile (32 keys x 64 dh): async DMA to LDS ----
        for (int c = tid; c < 256; c += 128) {       // 256 x 16B chunks
            int row = c >> 3, off = (c & 7) * 8;
            const unsigned short* src = Kh + base + (size_t)(kt + row) * DHc + off;
#if HAVE_ASYNC_LDS
            async_copy16(src, &lK[row * 64 + off]);
#else
            ushort4 a = ((const ushort4*)src)[0], bq = ((const ushort4*)src)[1];
            *(ushort4*)&lK[row * 64 + off]     = a;
            *(ushort4*)&lK[row * 64 + off + 4] = bq;
#endif
            if (kt + 32 < kEndBlk) __builtin_prefetch(src + 32 * DHc, 0, 1);
        }
        // ---- stage V tile transposed -> lVT[dh][key] ----
        for (int c = tid; c < 256; c += 128) {
            int key = c >> 3, d0 = (c & 7) * 8;
            const unsigned short* src = Vh + base + (size_t)(kt + key) * DHc + d0;
            ushort4 a = ((const ushort4*)src)[0], b4 = ((const ushort4*)src)[1];
            unsigned short tmp[8] = {a.x, a.y, a.z, a.w, b4.x, b4.y, b4.z, b4.w};
            if (kt + 32 < kEndBlk) __builtin_prefetch(src + 32 * DHc, 0, 1);
#pragma unroll
            for (int j = 0; j < 8; ++j)
                lVT[(d0 + j) * 32 + key] = tmp[j];
        }
#if HAVE_ASYNC_LDS
        asm volatile("s_wait_asynccnt 0" ::: "memory");
#endif
        __syncthreads();

        if (kt < q0 + 16) {   // wave-uniform causal skip
            // ---- scores S = Q * K^T for 32 keys (two 16x16 tiles) ----
            v8f s[2] = {};
#pragma unroll
            for (int c = 0; c < 2; ++c) {
#pragma unroll
                for (int f = 0; f < 2; ++f) {
                    ABFrag kb;
#pragma unroll
                    for (int r = 0; r < 8; ++r) {
                        int dh = kB(r, half) + 32 * f;
                        kb.u[r] = *(const unsigned int*)
                            &lK[(c * 16 + l16) * 64 + dh];
                    }
                    s[c] = __builtin_amdgcn_wmma_f32_16x16x32_bf16(
                        false, qa[f].v, false, kb.v, (short)0, s[c],
                        false, false);
                }
            }

            // ---- scale + causal mask ----
#pragma unroll
            for (int c = 0; c < 2; ++c) {
                int key = kt + c * 16 + l16;
#pragma unroll
                for (int r = 0; r < 8; ++r) {
                    int qrow = q0 + r + 8 * half;
                    float v = s[c][r] * scale;
                    s[c][r] = (key <= qrow) ? v : -1e30f;
                }
            }

            // ---- online softmax (row stats live in each 16-lane half) ----
#pragma unroll
            for (int r = 0; r < 8; ++r) {
                float t = fmaxf(s[0][r], s[1][r]);
#pragma unroll
                for (int m = 8; m >= 1; m >>= 1)
                    t = fmaxf(t, __shfl_xor(t, m, 32));
                float mNew = fmaxf(rowMax[r], t);
                float corr = __expf(rowMax[r] - mNew);
                float p0 = __expf(s[0][r] - mNew);
                float p1 = __expf(s[1][r] - mNew);
                s[0][r] = p0; s[1][r] = p1;
                float ps = p0 + p1;
#pragma unroll
                for (int m = 8; m >= 1; m >>= 1)
                    ps += __shfl_xor(ps, m, 32);
                rowSum[r] = rowSum[r] * corr + ps;
                rowMax[r] = mNew;
#pragma unroll
                for (int t2 = 0; t2 < 4; ++t2) o[t2][r] *= corr;
            }

            // ---- re-layout P (C-layout f32) -> A-layout bf16 via LDS ----
            unsigned short* P = lP[wave];
#pragma unroll
            for (int c = 0; c < 2; ++c)
#pragma unroll
                for (int r = 0; r < 8; ++r)
                    P[(r + 8 * half) * 32 + c * 16 + l16] = f2bf(s[c][r]);
            asm volatile("s_wait_dscnt 0" ::: "memory");   // per-wave LDS RAW
            ABFrag pa;
#pragma unroll
            for (int v = 0; v < 8; ++v)
                pa.u[v] = *(const unsigned int*)&P[l16 * 32 + kA(v, half)];

            // ---- O += P(16x32) * V(32x64), V frags from transposed LDS ----
#pragma unroll
            for (int t2 = 0; t2 < 4; ++t2) {
                ABFrag vb;
#pragma unroll
                for (int r = 0; r < 8; ++r)
                    vb.u[r] = *(const unsigned int*)
                        &lVT[(t2 * 16 + l16) * 32 + kB(r, half)];
                o[t2] = __builtin_amdgcn_wmma_f32_16x16x32_bf16(
                    false, pa.v, false, vb.v, (short)0, o[t2], false, false);
            }
        }
        __syncthreads();
    }

    // ---- normalize and store context [B,T,H*DH] bf16 ----
    const int D_ = Hc * DHc;
#pragma unroll
    for (int t2 = 0; t2 < 4; ++t2) {
        int dh = t2 * 16 + l16;
#pragma unroll
        for (int r = 0; r < 8; ++r) {
            int qrow = q0 + r + 8 * half;
            float val = o[t2][r] / rowSum[r];
            ctx[((size_t)(b * T_ + qrow)) * D_ + h * DHc + dh] = f2bf(val);
        }
    }
}

// ---------------------------------------------------------------------------
extern "C" void kernel_launch(void* const* d_in, const int* in_sizes, int n_in,
                              void* d_out, int out_size, void* d_ws, size_t ws_size,
                              hipStream_t stream) {
    (void)in_sizes; (void)n_in; (void)out_size; (void)ws_size;

    constexpr int B = 2, T = 2048, D = 1024, H = 16, DH = 64;
    constexpr int M = B * T;                            // 4096
    constexpr size_t HEADSZ = (size_t)B * H * T * DH;   // 4M elements

    const float* q  = (const float*)d_in[0];
    const float* k  = (const float*)d_in[1];
    const float* v  = (const float*)d_in[2];
    const float* Wq = (const float*)d_in[3];
    const float* bq = (const float*)d_in[4];
    const float* Wk = (const float*)d_in[5];
    const float* bk = (const float*)d_in[6];
    const float* Wv = (const float*)d_in[7];
    const float* bv = (const float*)d_in[8];
    const float* Wo = (const float*)d_in[9];
    const float* bo = (const float*)d_in[10];

    unsigned short* Qh  = (unsigned short*)d_ws;
    unsigned short* Kh  = Qh + HEADSZ;
    unsigned short* Vh  = Kh + HEADSZ;
    unsigned short* ctx = Vh + HEADSZ;        // [B,T,D] bf16

    dim3 gGemm(D / 128, M / 64);
    dim3 blk(128);

    // Q/K/V projections -> bf16 head-split [B,H,T,DH]
    gemm_bias_kernel<0, 1><<<gGemm, blk, 0, stream>>>(q, Wq, bq, Qh, M, D, D, T, H, DH);
    gemm_bias_kernel<0, 1><<<gGemm, blk, 0, stream>>>(k, Wk, bk, Kh, M, D, D, T, H, DH);
    gemm_bias_kernel<0, 1><<<gGemm, blk, 0, stream>>>(v, Wv, bv, Vh, M, D, D, T, H, DH);

    // causal flash attention -> bf16 context [B,T,D]
    dim3 gAttn(T / 64, B * H);
    attn_kernel<<<gAttn, blk, 0, stream>>>(Qh, Kh, Vh, ctx, T, H, DH);

    // output projection -> f32 [B,T,D]
    gemm_bias_kernel<1, 0><<<gGemm, blk, 0, stream>>>(ctx, Wo, bo, (float*)d_out, M, D, D, T, H, DH);
}
